// MHA_87170656240119
// MI455X (gfx1250) — matrix operand order repaired and checked by reading, not compiled, not run on hardware
//
#include <hip/hip_runtime.h>

#define BB 2
#define SS 2048
#define DD 1024
#define HH 16
#define HD 64
#define SM_SCALE 0.125f /* 1/sqrt(64) */

#define MBLK 128
#define NBLK 128
#define KSTEP 32

typedef __attribute__((ext_vector_type(16))) __bf16 v16bf;
typedef __attribute__((ext_vector_type(8)))  float  v8f;

__device__ __forceinline__ void storeC(float* p, float v) { *p = v; }
__device__ __forceinline__ void storeC(__bf16* p, float v) { *p = (__bf16)v; }

// ---------------------------------------------------------------------------
// Elementwise f32 -> bf16 (n % 4 == 0).
// ---------------------------------------------------------------------------
__global__ void cvt_f32_bf16(const float* __restrict__ in,
                             __bf16* __restrict__ out, int n) {
  int i = (blockIdx.x * blockDim.x + threadIdx.x) * 4;
  if (i < n) {
    float4 v = *(const float4*)(in + i);
    out[i]     = (__bf16)v.x;
    out[i + 1] = (__bf16)v.y;
    out[i + 2] = (__bf16)v.z;
    out[i + 3] = (__bf16)v.w;
  }
}

// ---------------------------------------------------------------------------
// Fold the head-sum into the K/V projection weights; emit bf16 weights.
// ---------------------------------------------------------------------------
__global__ void reduce_kv_weights(const float* __restrict__ Wqkv,
                                  const float* __restrict__ bqkv,
                                  __bf16* __restrict__ Wred,
                                  float* __restrict__ bred) {
  int idx = blockIdx.x * blockDim.x + threadIdx.x;
  if (idx < 128 * DD) {
    int j = idx / DD;
    int k = idx % DD;
    int base = (j < 64) ? DD : 2 * DD;
    int jj = j & 63;
    float s = 0.f;
#pragma unroll
    for (int h = 0; h < HH; ++h)
      s += Wqkv[(size_t)(base + h * HD + jj) * DD + k];
    Wred[(size_t)j * DD + k] = (__bf16)s;
  }
  if (idx < 128) {
    int j = idx;
    int base = (j < 64) ? DD : 2 * DD;
    int jj = j & 63;
    float s = 0.f;
#pragma unroll
    for (int h = 0; h < HH; ++h) s += bqkv[base + h * HD + jj];
    bred[j] = s;
  }
}

// ---------------------------------------------------------------------------
// LDS-tiled bf16 WMMA GEMM:  C[M x N] = A[M x K] * W[N x K]^T + bias[N]
// 128x128 block tile, 8 waves x (32x64 register tile), K stepped by 32.
// bf16 tiles staged via async B128 loads (ASYNCcnt), double-buffered;
// zero converts in the inner loop.  CT = float or __bf16 output.
// ---------------------------------------------------------------------------
template <typename CT>
__global__ void __launch_bounds__(256)
wmma_gemm_tiled(const __bf16* __restrict__ A, const __bf16* __restrict__ W,
                const float* __restrict__ bias, CT* __restrict__ C,
                int M, int N, int K) {
  __shared__ __bf16 As[2][MBLK * KSTEP];
  __shared__ __bf16 Ws[2][NBLK * KSTEP];

  const int t = threadIdx.x;
  const int lane = t & 31;
  const int wave = t >> 5;
  const int nl = lane & 15;
  const int half = lane >> 4;
  const int nbn = N / NBLK;
  const int m0 = (blockIdx.x / nbn) * MBLK;
  const int n0 = (blockIdx.x % nbn) * NBLK;

  // Stage one bf16 K-tile of A (128x32) and W (128x32): 512 B128 chunks per
  // matrix, 2 per thread: chunk -> (row, 8-elem group).
  auto stage = [&](int buf, int k0) {
#pragma unroll
    for (int j = 0; j < 2; ++j) {
      int chunk = j * 256 + t;   // 0..511
      int row = chunk >> 2;      // 0..127
      int qq = chunk & 3;        // 16-byte group within the 64-byte row
      unsigned ldsA = (unsigned)(size_t)&As[buf][row * KSTEP + qq * 8];
      unsigned long long ga =
          (unsigned long long)(A + (size_t)(m0 + row) * K + k0 + qq * 8);
      asm volatile("global_load_async_to_lds_b128 %0, %1, off"
                   :: "v"(ldsA), "v"(ga) : "memory");
      unsigned ldsW = (unsigned)(size_t)&Ws[buf][row * KSTEP + qq * 8];
      unsigned long long gw =
          (unsigned long long)(W + (size_t)(n0 + row) * K + k0 + qq * 8);
      asm volatile("global_load_async_to_lds_b128 %0, %1, off"
                   :: "v"(ldsW), "v"(gw) : "memory");
    }
  };

  const int wm = (wave & 3) * 32;   // M sub-tile: 0/32/64/96
  const int wn = (wave >> 2) * 64;  // N sub-tile: 0/64
  v8f acc[2][4] = {};

  stage(0, 0);
  const int nk = K / KSTEP;
  for (int ks = 0; ks < nk; ++ks) {
    if (ks + 1 < nk) {
      stage((ks + 1) & 1, (ks + 1) * KSTEP);
      // 4 loads of the next tile may stay in flight; current tile must land.
      asm volatile("s_wait_asynccnt 0x4" ::: "memory");
    } else {
      asm volatile("s_wait_asynccnt 0x0" ::: "memory");
    }
    __syncthreads();

    const __bf16* as = As[ks & 1];
    const __bf16* ws = Ws[ks & 1];
    v16bf afr[2], bfr[4];
#pragma unroll
    for (int mt = 0; mt < 2; ++mt) {
      const __bf16* ar = as + (wm + mt * 16 + nl) * KSTEP;
#pragma unroll
      for (int i = 0; i < 8; ++i) {
        afr[mt][i]     = ar[half * 8 + i];
        afr[mt][8 + i] = ar[16 + half * 8 + i];
      }
    }
#pragma unroll
    for (int nt = 0; nt < 4; ++nt) {
      const __bf16* wr = ws + (wn + nt * 16 + nl) * KSTEP;
#pragma unroll
      for (int i = 0; i < 8; ++i) {
        bfr[nt][i]     = wr[half * 8 + i];
        bfr[nt][8 + i] = wr[16 + half * 8 + i];
      }
    }
#pragma unroll
    for (int mt = 0; mt < 2; ++mt)
#pragma unroll
      for (int nt = 0; nt < 4; ++nt)
        acc[mt][nt] = __builtin_amdgcn_wmma_f32_16x16x32_bf16(
            false, afr[mt], false, bfr[nt], (short)0, acc[mt][nt], false, false);
    __syncthreads();
  }

  // Epilogue: bias add + store (f32 or bf16).
#pragma unroll
  for (int mt = 0; mt < 2; ++mt) {
#pragma unroll
    for (int nt = 0; nt < 4; ++nt) {
      int ncol = n0 + wn + nt * 16 + nl;
      float bn = bias[ncol];
#pragma unroll
      for (int r = 0; r < 8; ++r) {
        int m = m0 + wm + mt * 16 + half * 8 + r;
        storeC(&C[(size_t)m * N + ncol], acc[mt][nt][r] + bn);
      }
    }
  }
}

// ---------------------------------------------------------------------------
// Flash-attention over head-summed K/V (all operands bf16, softmax in fp32).
// Q: (B,S,1024) bf16.  KV: (B,S,128) bf16, cols 0..63=K_sum, 64..127=V_sum.
// O: (B,S,1024) bf16 attn output.
// ---------------------------------------------------------------------------
__global__ void __launch_bounds__(128)
flash_attn_headsum(const __bf16* __restrict__ Q, const __bf16* __restrict__ KV,
                   __bf16* __restrict__ O) {
  __shared__ __bf16 pbuf[4][16 * 32];  // P staging: 16x32 bf16 per wave

  const int lane = threadIdx.x & 31;
  const int wv = threadIdx.x >> 5;
  const int S16 = SS / 16;
  int w = blockIdx.x * 4 + wv;
  const int qt = w % S16; w /= S16;
  const int h  = w % HH;  w /= HH;
  const int b  = w;
  const int q0 = qt * 16;
  const int half = lane >> 4;
  const int nl = lane & 15;

  // Q A-fragments: 16 rows x 64 features -> two K=32 fragments, loaded once.
  v16bf aq0, aq1;
  {
    const __bf16* qrow = Q + ((size_t)(b * SS) + q0 + nl) * DD + h * HD;
#pragma unroll
    for (int i = 0; i < 16; ++i) {
      aq0[i] = qrow[half * 16 + i];
      aq1[i] = qrow[32 + half * 16 + i];
    }
  }

  float m_i[8], l_i[8], alpha[8];
  v8f acc[4] = {};
#pragma unroll
  for (int r = 0; r < 8; ++r) { m_i[r] = -1e30f; l_i[r] = 0.f; }

  const __bf16* kvb = KV + (size_t)(b * SS) * 128;
  const int nsteps = (q0 + 16 + 31) >> 5;  // 32 keys per step, causal bound
  __bf16* pb = pbuf[wv];

  for (int it = 0; it < nsteps; ++it) {
    const int t0 = it * 32;

    // ---- scores: S = Q(16x64) @ Ksum^T(64x16), two 16-wide key sub-tiles
    v8f s0 = {}, s1 = {};
    {
      v16bf bk0, bk1;
      const __bf16* kr = kvb + (size_t)(t0 + nl) * 128;
#pragma unroll
      for (int i = 0; i < 16; ++i) {
        bk0[i] = kr[half * 16 + i];
        bk1[i] = kr[32 + half * 16 + i];
      }
      s0 = __builtin_amdgcn_wmma_f32_16x16x32_bf16(false, aq0, false, bk0,
                                                   (short)0, s0, false, false);
      s0 = __builtin_amdgcn_wmma_f32_16x16x32_bf16(false, aq1, false, bk1,
                                                   (short)0, s0, false, false);
      const __bf16* kr1 = kvb + (size_t)(t0 + 16 + nl) * 128;
#pragma unroll
      for (int i = 0; i < 16; ++i) {
        bk0[i] = kr1[half * 16 + i];
        bk1[i] = kr1[32 + half * 16 + i];
      }
      s1 = __builtin_amdgcn_wmma_f32_16x16x32_bf16(false, aq0, false, bk0,
                                                   (short)0, s1, false, false);
      s1 = __builtin_amdgcn_wmma_f32_16x16x32_bf16(false, aq1, false, bk1,
                                                   (short)0, s1, false, false);
    }

    // ---- scale + causal mask + online softmax (row stats in fp32)
#pragma unroll
    for (int r = 0; r < 8; ++r) {
      const int srow = q0 + half * 8 + r;
      float v0 = s0[r] * SM_SCALE;
      float v1 = s1[r] * SM_SCALE;
      if (t0 + nl > srow) v0 = -1e30f;
      if (t0 + 16 + nl > srow) v1 = -1e30f;

      float mx = fmaxf(v0, v1);
      mx = fmaxf(mx, __shfl_xor(mx, 1, 32));
      mx = fmaxf(mx, __shfl_xor(mx, 2, 32));
      mx = fmaxf(mx, __shfl_xor(mx, 4, 32));
      mx = fmaxf(mx, __shfl_xor(mx, 8, 32));
      float mn = fmaxf(m_i[r], mx);
      alpha[r] = __expf(m_i[r] - mn);
      m_i[r] = mn;
      float p0 = __expf(v0 - mn);
      float p1 = __expf(v1 - mn);
      s0[r] = p0;
      s1[r] = p1;
      float rs = p0 + p1;
      rs += __shfl_xor(rs, 1, 32);
      rs += __shfl_xor(rs, 2, 32);
      rs += __shfl_xor(rs, 4, 32);
      rs += __shfl_xor(rs, 8, 32);
      l_i[r] = l_i[r] * alpha[r] + rs;
    }
#pragma unroll
    for (int c = 0; c < 4; ++c)
#pragma unroll
      for (int r = 0; r < 8; ++r) acc[c][r] *= alpha[r];

    // ---- re-layout P (C-frag) -> A-frag via LDS (16x32 bf16, wave-local)
#pragma unroll
    for (int r = 0; r < 8; ++r) {
      int m = half * 8 + r;
      pb[m * 32 + nl]      = (__bf16)s0[r];
      pb[m * 32 + 16 + nl] = (__bf16)s1[r];
    }
    asm volatile("s_wait_dscnt 0" ::: "memory");
    v16bf ap;
#pragma unroll
    for (int i = 0; i < 16; ++i) ap[i] = pb[nl * 32 + half * 16 + i];

    // ---- O += P(16x32) @ Vsum(32x64), four 16-col output chunks
#pragma unroll
    for (int c = 0; c < 4; ++c) {
      v16bf bv;
#pragma unroll
      for (int i = 0; i < 16; ++i)
        bv[i] = kvb[(size_t)(t0 + half * 16 + i) * 128 + 64 + c * 16 + nl];
      acc[c] = __builtin_amdgcn_wmma_f32_16x16x32_bf16(false, ap, false, bv,
                                                       (short)0, acc[c], false, false);
    }
    asm volatile("s_wait_dscnt 0" ::: "memory");  // loads done before re-store
  }

  // ---- finalize: divide by row sum, write attn_out (B,S,H*64) as bf16
#pragma unroll
  for (int r = 0; r < 8; ++r) {
    float inv = 1.f / l_i[r];
    int m = q0 + half * 8 + r;
    __bf16* orow = O + ((size_t)(b * SS) + m) * DD + h * HD;
#pragma unroll
    for (int c = 0; c < 4; ++c) orow[c * 16 + nl] = (__bf16)(acc[c][r] * inv);
  }
}

// ---------------------------------------------------------------------------
extern "C" void kernel_launch(void* const* d_in, const int* in_sizes, int n_in,
                              void* d_out, int out_size, void* d_ws, size_t ws_size,
                              hipStream_t stream) {
  (void)in_sizes; (void)n_in; (void)out_size; (void)ws_size;
  const float* x    = (const float*)d_in[0];
  const float* Wqkv = (const float*)d_in[1];
  const float* bqkv = (const float*)d_in[2];
  const float* Wout = (const float*)d_in[3];
  const float* bout = (const float*)d_in[4];
  float* out = (float*)d_out;

  char* ws = (char*)d_ws;
  __bf16* Wredb = (__bf16*)ws; ws += (size_t)128 * DD * sizeof(__bf16);
  float*  bred  = (float*)ws;  ws += 128 * sizeof(float);
  __bf16* xb    = (__bf16*)ws; ws += (size_t)BB * SS * DD * sizeof(__bf16);
  __bf16* Wqb   = (__bf16*)ws; ws += (size_t)DD * DD * sizeof(__bf16);
  __bf16* Woutb = (__bf16*)ws; ws += (size_t)DD * DD * sizeof(__bf16);
  __bf16* Qb    = (__bf16*)ws; ws += (size_t)BB * SS * DD * sizeof(__bf16);
  __bf16* KVb   = (__bf16*)ws; ws += (size_t)BB * SS * 128 * sizeof(__bf16);
  __bf16* AOb   = (__bf16*)ws;

  const int M = BB * SS;

  // Edge conversions to bf16 (once) + weight head-sum fold.
  cvt_f32_bf16<<<(M * DD / 4 + 255) / 256, 256, 0, stream>>>(x, xb, M * DD);
  cvt_f32_bf16<<<(DD * DD / 4 + 255) / 256, 256, 0, stream>>>(Wqkv, Wqb, DD * DD);
  cvt_f32_bf16<<<(DD * DD / 4 + 255) / 256, 256, 0, stream>>>(Wout, Woutb, DD * DD);
  reduce_kv_weights<<<(128 * DD + 255) / 256, 256, 0, stream>>>(Wqkv, bqkv,
                                                                Wredb, bred);
  // Q = x @ Wq^T + bq  (bf16 out)
  wmma_gemm_tiled<__bf16><<<(M / MBLK) * (DD / NBLK), 256, 0, stream>>>(
      xb, Wqb, bqkv, Qb, M, DD, DD);
  // KV = x @ Wred^T + bred  (128 cols: K_sum | V_sum, bf16 out)
  wmma_gemm_tiled<__bf16><<<(M / MBLK) * (128 / NBLK), 256, 0, stream>>>(
      xb, Wredb, bred, KVb, M, 128, DD);
  // attention (bf16 in/out, fp32 softmax)
  flash_attn_headsum<<<BB * HH * (SS / 16) / 4, 128, 0, stream>>>(Qb, KVb, AOb);
  // out = attn_out @ Wout^T + bout  (f32 out)
  wmma_gemm_tiled<float><<<(M / MBLK) * (DD / NBLK), 256, 0, stream>>>(
      AOb, Woutb, bout, out, M, DD, DD);
}
